// GPRGNN_14370960572520
// MI455X (gfx1250) — compile-verified, matrix operand order checked by reading, not run on hardware
//
#include <hip/hip_runtime.h>
#include <math.h>

// ---------------- problem constants (match reference) ----------------
#define NNODES   100000
#define FIN      500
#define KPAD     512      // FIN padded to multiple of 32 for WMMA K-steps
#define HID      256
#define FOUT     40
#define FPAD     48       // FOUT padded to 3 WMMA N-tiles; cols 40..47 == 0
#define KSTEPS   10

typedef __attribute__((ext_vector_type(16))) __bf16 v16bf;
typedef __attribute__((ext_vector_type(8)))  __bf16 v8bf;
typedef __attribute__((ext_vector_type(8)))  float  v8f;

// Load one 16x32 bf16 WMMA fragment (A, or B via pre-transposed weight):
// lane<16 needs K[0..7] and K[16..23] of its row, lane>=16 needs K[8..15]
// and K[24..31]. Both are contiguous 16B chunks -> two global_load_b128.
__device__ __forceinline__ v16bf frag_ld(const __bf16* p) {
    v8bf a = *(const v8bf*)(p);
    v8bf b = *(const v8bf*)(p + 16);
    v16bf f;
#pragma unroll
    for (int j = 0; j < 8; ++j) { f[j] = a[j]; f[8 + j] = b[j]; }
    return f;
}

// ---------------- prep: bf16 conversions (padded / transposed) ----------------
__global__ void cvt_x_kernel(const float* __restrict__ x, __bf16* __restrict__ xb) {
    size_t idx = (size_t)blockIdx.x * blockDim.x + threadIdx.x;
    if (idx >= (size_t)NNODES * KPAD) return;
    int m = (int)(idx >> 9), k = (int)(idx & (KPAD - 1));
    float v = (k < FIN) ? x[(size_t)m * FIN + k] : 0.0f;
    xb[idx] = (__bf16)v;
}

__global__ void cvt_w1t_kernel(const float* __restrict__ w1, __bf16* __restrict__ w1t) {
    int idx = blockIdx.x * blockDim.x + threadIdx.x;
    if (idx >= HID * KPAD) return;
    int n = idx >> 9, k = idx & (KPAD - 1);
    float v = (k < FIN) ? w1[(size_t)k * HID + n] : 0.0f;
    w1t[idx] = (__bf16)v;
}

__global__ void cvt_w2t_kernel(const float* __restrict__ w2, __bf16* __restrict__ w2t) {
    int idx = blockIdx.x * blockDim.x + threadIdx.x;
    if (idx >= FPAD * HID) return;
    int n = idx >> 8, k = idx & (HID - 1);
    float v = (n < FOUT) ? w2[(size_t)k * FOUT + n] : 0.0f;
    w2t[idx] = (__bf16)v;
}

// ---------------- CSR build (by destination) ----------------
__global__ void zero_deg_kernel(int* __restrict__ deg) {
    int i = blockIdx.x * blockDim.x + threadIdx.x;
    if (i < NNODES) deg[i] = 0;
}

__global__ void count_deg_kernel(const int* __restrict__ dst, int* __restrict__ deg, int E) {
    int e = blockIdx.x * blockDim.x + threadIdx.x;
    if (e < E) atomicAdd(&deg[dst[e]], 1);
}

// single-block inclusive scan -> exclusive offsets (100k ints: trivial work)
__global__ __launch_bounds__(1024) void scan_kernel(const int* __restrict__ deg,
                                                    int* __restrict__ offs) {
    __shared__ int sm[1024];
    __shared__ int carry;
    const int t = threadIdx.x;
    if (t == 0) carry = 0;
    __syncthreads();
    for (int base = 0; base < NNODES; base += 1024) {
        int i = base + t;
        int v = (i < NNODES) ? deg[i] : 0;
        sm[t] = v;
        __syncthreads();
#pragma unroll
        for (int off = 1; off < 1024; off <<= 1) {
            int add = (t >= off) ? sm[t - off] : 0;
            __syncthreads();
            sm[t] += add;
            __syncthreads();
        }
        int inc = sm[t];
        int c = carry;
        if (i < NNODES) offs[i + 1] = c + inc;
        __syncthreads();
        if (t == 1023) carry = c + inc;
        __syncthreads();
    }
    if (t == 0) offs[0] = 0;
}

// dinv from offsets (+1 self-loop), and init cursor = offs
__global__ void dinv_cursor_kernel(const int* __restrict__ offs, float* __restrict__ dinv,
                                   int* __restrict__ cursor) {
    int i = blockIdx.x * blockDim.x + threadIdx.x;
    if (i >= NNODES) return;
    int beg = offs[i];
    dinv[i] = rsqrtf((float)(offs[i + 1] - beg + 1));
    cursor[i] = beg;
}

// scatter edges into CSR slots; precompute edge weight dinv[s]*dinv[d]
__global__ void fill_csr_kernel(const int* __restrict__ src, const int* __restrict__ dst,
                                const float* __restrict__ dinv, int* __restrict__ cursor,
                                int* __restrict__ csr_src, float* __restrict__ csr_w, int E) {
    int e = blockIdx.x * blockDim.x + threadIdx.x;
    if (e >= E) return;
    int s = src[e], d = dst[e];
    int slot = atomicAdd(&cursor[d], 1);
    csr_src[slot] = s;
    csr_w[slot] = dinv[s] * dinv[d];
}

// ---------------- GEMM1: h1 = relu(xb @ w1 + b1), bf16 out ----------------
// grid.x = NNODES/32 ; block = 256 (8 waves); wave w covers 32 rows x cols [32w,32w+32)
__global__ __launch_bounds__(256) void gemm1_kernel(
        const __bf16* __restrict__ xb, const __bf16* __restrict__ w1t,
        const float* __restrict__ b1, __bf16* __restrict__ h1b) {
    const int lane = threadIdx.x & 31;
    const int wave = threadIdx.x >> 5;
    const int m0 = blockIdx.x * 32;
    const int n0 = wave * 32;
    const int lr = lane & 15;
    const int hi = lane >> 4;           // 0: K 0-7/16-23 ; 1: K 8-15/24-31

    const __bf16* arow0 = xb  + (size_t)(m0 + lr) * KPAD + hi * 8;
    const __bf16* arow1 = arow0 + (size_t)16 * KPAD;
    const __bf16* brow0 = w1t + (size_t)(n0 + lr) * KPAD + hi * 8;
    const __bf16* brow1 = brow0 + (size_t)16 * KPAD;

    v8f acc00 = {}; v8f acc01 = {}; v8f acc10 = {}; v8f acc11 = {};
#pragma unroll 2
    for (int kk = 0; kk < KPAD; kk += 32) {
        v16bf a0 = frag_ld(arow0 + kk);
        v16bf a1 = frag_ld(arow1 + kk);
        v16bf bb0 = frag_ld(brow0 + kk);
        v16bf bb1 = frag_ld(brow1 + kk);
        acc00 = __builtin_amdgcn_wmma_f32_16x16x32_bf16(false, a0, false, bb0,
                                                        (short)0, acc00, false, false);
        acc01 = __builtin_amdgcn_wmma_f32_16x16x32_bf16(false, a0, false, bb1,
                                                        (short)0, acc01, false, false);
        acc10 = __builtin_amdgcn_wmma_f32_16x16x32_bf16(false, a1, false, bb0,
                                                        (short)0, acc10, false, false);
        acc11 = __builtin_amdgcn_wmma_f32_16x16x32_bf16(false, a1, false, bb1,
                                                        (short)0, acc11, false, false);
    }
    const int col = n0 + lr;
    const float bias0 = b1[col], bias1 = b1[col + 16];
#pragma unroll
    for (int r = 0; r < 8; ++r) {
        int row0 = m0 + r + hi * 8;     // C layout: VGPR r -> M=r (lanes<16) / 8+r
        int row1 = row0 + 16;
        float v;
        v = acc00[r] + bias0; v = v > 0.0f ? v : 0.0f; h1b[(size_t)row0 * HID + col]      = (__bf16)v;
        v = acc01[r] + bias1; v = v > 0.0f ? v : 0.0f; h1b[(size_t)row0 * HID + col + 16] = (__bf16)v;
        v = acc10[r] + bias0; v = v > 0.0f ? v : 0.0f; h1b[(size_t)row1 * HID + col]      = (__bf16)v;
        v = acc11[r] + bias1; v = v > 0.0f ? v : 0.0f; h1b[(size_t)row1 * HID + col + 16] = (__bf16)v;
    }
}

// ---------------- GEMM2: h = h1b @ w2 + b2  (+ hidden = temp[0]*h) ----------------
// grid.x = NNODES/32 ; block = 192 (6 waves): wave = mt*3 + nt
__global__ __launch_bounds__(192) void gemm2_kernel(
        const __bf16* __restrict__ h1b, const __bf16* __restrict__ w2t,
        const float* __restrict__ b2, const float* __restrict__ temp,
        float* __restrict__ hbuf, float* __restrict__ hidden) {
    const int lane = threadIdx.x & 31;
    const int wave = threadIdx.x >> 5;
    const int mt = wave / 3, nt = wave % 3;
    const int m0 = blockIdx.x * 32 + mt * 16;
    const int n0 = nt * 16;
    const int lr = lane & 15;
    const int hi = lane >> 4;

    const __bf16* arow = h1b + (size_t)(m0 + lr) * HID + hi * 8;
    const __bf16* brow = w2t + (size_t)(n0 + lr) * HID + hi * 8;

    v8f acc = {};
#pragma unroll
    for (int kk = 0; kk < HID; kk += 32) {
        v16bf af = frag_ld(arow + kk);
        v16bf bf = frag_ld(brow + kk);
        acc = __builtin_amdgcn_wmma_f32_16x16x32_bf16(false, af, false, bf,
                                                      (short)0, acc, false, false);
    }
    const int col = n0 + lr;
    const float bias = (col < FOUT) ? b2[col] : 0.0f;   // padded cols stay exactly 0
    const float g0 = temp[0];
#pragma unroll
    for (int r = 0; r < 8; ++r) {
        int row = m0 + r + hi * 8;
        float v = acc[r] + bias;
        hbuf[(size_t)row * FPAD + col]   = v;
        hidden[(size_t)row * FPAD + col] = g0 * v;
    }
}

// ---------------- propagation: CSR gather, one wave per dst node ----------------
// nxt[d] = dinv[d]^2*h[d] + sum_e w_e * h[src_e] ;  hidden += temp[k] * nxt[d]
// h is L2-resident (19.2 MB); no atomics in the hot loop.
__global__ __launch_bounds__(256) void prop_gather_kernel(
        const float* __restrict__ h, const int* __restrict__ offs,
        const int* __restrict__ csr_src, const float* __restrict__ csr_w,
        const float* __restrict__ dinv, const float* __restrict__ temp, int k,
        float* __restrict__ nxt, float* __restrict__ hidden) {
    int row = blockIdx.x * 8 + (threadIdx.x >> 5);
    if (row >= NNODES) return;
    int lane = threadIdx.x & 31;
    const int beg = offs[row], end = offs[row + 1];
    const float di = dinv[row];
    const float wself = di * di;
    const float* hr = h + (size_t)row * FPAD;
    float a0 = wself * hr[lane];
    float a1 = (lane < 16) ? wself * hr[32 + lane] : 0.0f;
    for (int j = beg; j < end; ++j) {
        int s = csr_src[j];
        float w = csr_w[j];
        if (j + 1 < end)   // hide the data-dependent row-base latency
            __builtin_prefetch(h + (size_t)csr_src[j + 1] * FPAD, 0, 1);
        const float* hs = h + (size_t)s * FPAD;
        a0 += w * hs[lane];
        if (lane < 16) a1 += w * hs[32 + lane];
    }
    size_t o = (size_t)row * FPAD + lane;
    const float g = temp[k];
    nxt[o] = a0;
    hidden[o] += g * a0;
    if (lane < 16) {
        nxt[o + 32] = a1;
        hidden[o + 32] += g * a1;
    }
}

// ---------------- log_softmax over 40 classes, one wave per row ----------------
__global__ __launch_bounds__(256) void lsm_kernel(const float* __restrict__ hidden,
                                                  float* __restrict__ out) {
    int row = blockIdx.x * 8 + (threadIdx.x >> 5);
    if (row >= NNODES) return;
    int lane = threadIdx.x & 31;
    const float* hr = hidden + (size_t)row * FPAD;
    float v1 = hr[lane];
    float v2 = (lane < FOUT - 32) ? hr[32 + lane] : -INFINITY;
    float m = fmaxf(v1, v2);
#pragma unroll
    for (int o = 16; o > 0; o >>= 1) m = fmaxf(m, __shfl_xor(m, o, 32));
    float s = __expf(v1 - m) + ((lane < FOUT - 32) ? __expf(v2 - m) : 0.0f);
#pragma unroll
    for (int o = 16; o > 0; o >>= 1) s += __shfl_xor(s, o, 32);
    float ls = m + __logf(s);
    out[(size_t)row * FOUT + lane] = v1 - ls;
    if (lane < FOUT - 32) out[(size_t)row * FOUT + 32 + lane] = v2 - ls;
}

// ---------------- host side ----------------
extern "C" void kernel_launch(void* const* d_in, const int* in_sizes, int n_in,
                              void* d_out, int out_size, void* d_ws, size_t ws_size,
                              hipStream_t stream) {
    const float* x    = (const float*)d_in[0];
    const int*   ei   = (const int*)  d_in[1];
    const float* w1   = (const float*)d_in[2];
    const float* b1   = (const float*)d_in[3];
    const float* w2   = (const float*)d_in[4];
    const float* b2   = (const float*)d_in[5];
    const float* temp = (const float*)d_in[6];
    float* out = (float*)d_out;

    const int E = in_sizes[1] / 2;
    const int* srcp = ei;
    const int* dstp = ei + E;

    // workspace carve-out (256B aligned slabs)
    char* ws = (char*)d_ws;
    size_t off = 0;
    auto carve = [&](size_t bytes) -> void* {
        off = (off + 255) & ~(size_t)255;
        void* p = ws + off;
        off += bytes;
        return p;
    };
    __bf16* xb      = (__bf16*)carve((size_t)NNODES * KPAD * 2);   // 102.4 MB
    __bf16* w1t     = (__bf16*)carve((size_t)HID * KPAD * 2);      // 256 KB (transposed)
    __bf16* w2t     = (__bf16*)carve((size_t)FPAD * HID * 2);      // 24 KB  (transposed)
    __bf16* h1b     = (__bf16*)carve((size_t)NNODES * HID * 2);    // 51.2 MB
    float*  hA      = (float*) carve((size_t)NNODES * FPAD * 4);   // 19.2 MB
    float*  hB      = (float*) carve((size_t)NNODES * FPAD * 4);   // 19.2 MB
    float*  hid     = (float*) carve((size_t)NNODES * FPAD * 4);   // 19.2 MB
    int*    deg     = (int*)   carve((size_t)NNODES * 4);
    int*    offs    = (int*)   carve(((size_t)NNODES + 1) * 4);
    int*    cursor  = (int*)   carve((size_t)NNODES * 4);
    float*  dinv    = (float*) carve((size_t)NNODES * 4);
    int*    csr_src = (int*)   carve((size_t)E * 4);               // 6.4 MB
    float*  csr_w   = (float*) carve((size_t)E * 4);               // 6.4 MB

    // bf16 conversions
    {
        size_t tot = (size_t)NNODES * KPAD;
        cvt_x_kernel<<<(unsigned)((tot + 255) / 256), 256, 0, stream>>>(x, xb);
        cvt_w1t_kernel<<<(HID * KPAD + 255) / 256, 256, 0, stream>>>(w1, w1t);
        cvt_w2t_kernel<<<(FPAD * HID + 255) / 256, 256, 0, stream>>>(w2, w2t);
    }

    // CSR build (once per launch)
    zero_deg_kernel<<<(NNODES + 255) / 256, 256, 0, stream>>>(deg);
    count_deg_kernel<<<(E + 255) / 256, 256, 0, stream>>>(dstp, deg, E);
    scan_kernel<<<1, 1024, 0, stream>>>(deg, offs);
    dinv_cursor_kernel<<<(NNODES + 255) / 256, 256, 0, stream>>>(offs, dinv, cursor);
    fill_csr_kernel<<<(E + 255) / 256, 256, 0, stream>>>(srcp, dstp, dinv, cursor,
                                                         csr_src, csr_w, E);

    // MLP via WMMA
    gemm1_kernel<<<NNODES / 32, 256, 0, stream>>>(xb, w1t, b1, h1b);
    gemm2_kernel<<<NNODES / 32, 192, 0, stream>>>(h1b, w2t, b2, temp, hA, hid);

    // K propagation rounds (ping-pong hA/hB), atomic-free CSR gather
    float* cur = hA;
    float* nxt = hB;
    for (int k = 0; k < KSTEPS; ++k) {
        prop_gather_kernel<<<(NNODES + 7) / 8, 256, 0, stream>>>(
            cur, offs, csr_src, csr_w, dinv, temp, k + 1, nxt, hid);
        float* t = cur; cur = nxt; nxt = t;
    }

    // output
    lsm_kernel<<<(NNODES + 7) / 8, 256, 0, stream>>>(hid, out);
    (void)n_in; (void)out_size; (void)ws_size;
}
